// GraphSAGEModel_12927851561250
// MI455X (gfx1250) — matrix operand order repaired and checked by reading, not compiled, not run on hardware
//
#include <hip/hip_runtime.h>
#include <hip/hip_bf16.h>

// ---------------------------------------------------------------------------
// GraphSAGE forward for MI455X (gfx1250, wave32, WMMA).
//   x[N,64] -> BN -> mean-aggregate over E edges -> h = [agg|xn]@[W_l;W_r]+b_l
//   -> relu -> relu(h@W1+b1)@W2+b2 -> out[N,2]
// GEMMs use v_wmma_f32_16x16x32_bf16 (f32 accumulate). xn/agg stay L2-resident
// (25.6MB each << 192MB L2) so gather/scatter atomics resolve in L2.
// Weights are prepacked into exact WMMA B-fragment layout and staged in LDS;
// an explicit compiler memory barrier in the tile loop prevents LICM from
// hoisting 36 B fragments into registers and spilling to scratch (seen in the
// previous round's disassembly).
// ---------------------------------------------------------------------------

typedef __bf16  bf16;
typedef __attribute__((ext_vector_type(16))) __bf16        v16bf;
typedef __attribute__((ext_vector_type(8)))  float         v8f;
typedef __attribute__((ext_vector_type(4)))  unsigned int  uv4;

union FragBF { uv4 u[2]; v16bf v; };

#define D_FEAT 64
#define H_FEAT 128
#define BN_EPS 1e-5f
#define NFRAG  36                 // 32 frags for W'=[W_l;W_r], 4 for W1

// ---------------- zero fill ----------------
__global__ void zero_f32(float* __restrict__ p, long long n) {
    long long i = (long long)blockIdx.x * blockDim.x + threadIdx.x;
    long long stride = (long long)gridDim.x * blockDim.x;
    for (; i < n; i += stride) p[i] = 0.0f;
}

// ---------------- BN column stats (sum, sumsq) ----------------
__global__ void bn_stats(const float* __restrict__ x,
                         float* __restrict__ sums, float* __restrict__ sumsqs,
                         int N) {
    __shared__ float red[512];
    const int tid = threadIdx.x;
    const int c   = tid & 63;
    const int grp = tid >> 6;           // 4 row-groups per block
    float s = 0.0f, q = 0.0f;
    for (int r = blockIdx.x * 4 + grp; r < N; r += gridDim.x * 4) {
        float v = x[(size_t)r * D_FEAT + c];
        s += v; q += v * v;
    }
    red[tid] = s; red[256 + tid] = q;
    __syncthreads();
    if (tid < 64) {
        float ts = red[tid] + red[tid + 64] + red[tid + 128] + red[tid + 192];
        float tq = red[256 + tid] + red[320 + tid] + red[384 + tid] + red[448 + tid];
        atomicAdd(&sums[tid], ts);
        atomicAdd(&sumsqs[tid], tq);
    }
}

__global__ void bn_finalize(const float* __restrict__ sums,
                            const float* __restrict__ sumsqs,
                            const float* __restrict__ gamma,
                            const float* __restrict__ beta,
                            float* __restrict__ scale,
                            float* __restrict__ shift, int N) {
    int d = threadIdx.x;
    if (d >= D_FEAT) return;
    float inv_n = 1.0f / (float)N;
    float mean  = sums[d] * inv_n;
    float var   = sumsqs[d] * inv_n - mean * mean;
    float s     = gamma[d] * rsqrtf(var + BN_EPS);
    scale[d] = s;
    shift[d] = beta[d] - mean * s;
}

// ---------------- BN apply: xn -> bf16 into combined A at K in [64,128) ----
__global__ void bn_apply(const float* __restrict__ x,
                         const float* __restrict__ scale,
                         const float* __restrict__ shift,
                         bf16* __restrict__ acomb, long long total) {
    long long i = (long long)blockIdx.x * blockDim.x + threadIdx.x;
    long long stride = (long long)gridDim.x * blockDim.x;
    for (; i < total; i += stride) {
        long long row = i >> 6;
        int c = (int)(i & 63);
        float v = x[i] * scale[c] + shift[c];
        acomb[row * (long long)H_FEAT + D_FEAT + c] = (bf16)v;
    }
}

// ---------------- edge scatter: agg[dst] += xn[src], deg[dst] += 1 --------
__global__ void edge_scatter(const long long* __restrict__ ei,
                             const bf16* __restrict__ acomb,
                             float* __restrict__ agg,
                             float* __restrict__ deg, int E) {
    long long gid = (long long)blockIdx.x * blockDim.x + threadIdx.x;
    int j = (int)(gid & 63);
    long long e = gid >> 6;
    long long stride = ((long long)gridDim.x * blockDim.x) >> 6;
    for (; e < E; e += stride) {
        int src = (int)ei[e];
        int dst = (int)ei[(long long)E + e];
        float v = (float)acomb[(long long)src * H_FEAT + D_FEAT + j];
        atomicAdd(&agg[(long long)dst * D_FEAT + j], v);
        if (j == 0) atomicAdd(&deg[dst], 1.0f);
    }
}

// ---------------- normalize: agg/deg -> bf16 into combined A at K in [0,64)
__global__ void agg_normalize(const float* __restrict__ agg,
                              const float* __restrict__ deg,
                              bf16* __restrict__ acomb, long long total) {
    long long i = (long long)blockIdx.x * blockDim.x + threadIdx.x;
    long long stride = (long long)gridDim.x * blockDim.x;
    for (; i < total; i += stride) {
        long long row = i >> 6;
        int c = (int)(i & 63);
        float d = deg[row];
        d = d > 1.0f ? d : 1.0f;
        acomb[row * (long long)H_FEAT + c] = (bf16)(agg[i] / d);
    }
}

// ---------------- prepack W'=[W_l;W_r] (128x128) and W1 (128x16) into
// per-lane WMMA B-fragment order: frag f, lane l, element v:
//   k = ks*32 + (l>>4)*16 + v ; n = base_n + (l&15)
// frags 0..31: f = nt*4+ks for W' ; frags 32..35: ks for W1.
__global__ void pack_w(const float* __restrict__ Wl,
                       const float* __restrict__ Wr,
                       const float* __restrict__ W1,
                       bf16* __restrict__ wpack) {
    int gid = blockIdx.x * blockDim.x + threadIdx.x;
    const int total = NFRAG * 32 * 16;
    if (gid >= total) return;
    int v    = gid & 15;
    int lane = (gid >> 4) & 31;
    int f    = gid >> 9;
    int half = lane >> 4;
    int n    = lane & 15;
    float val;
    if (f < 32) {
        int nt = f >> 2, ks = f & 3;
        int k  = ks * 32 + half * 16 + v;        // 0..127
        int col = nt * 16 + n;                   // 0..127
        val = (k < 64) ? Wl[k * H_FEAT + col]
                       : Wr[(k - 64) * H_FEAT + col];
    } else {
        int ks = f - 32;
        int k  = ks * 32 + half * 16 + v;        // 0..127
        val = W1[k * 16 + n];
    }
    wpack[gid] = (bf16)val;
}

// ---------------- fused WMMA GEMM + head ----------------------------------
// One wave handles one 16-node tile. 8 waves / block. Weights live in LDS.
__global__ __launch_bounds__(256) void
gemm_fused(const bf16* __restrict__ acomb,
           const bf16* __restrict__ wpack,
           const float* __restrict__ b_l,
           const float* __restrict__ b1,
           const float* __restrict__ W2,
           const float* __restrict__ b2,
           float* __restrict__ out,
           int Nnodes, int numTiles) {
    __shared__ bf16 hlds[8 * 16 * H_FEAT];       // 32KB: per-wave h buffers
    __shared__ bf16 wlds[NFRAG * 32 * 16];       // 36KB: B fragments

    // cooperative stage of all weight fragments into LDS (once per block)
    {
        const int nvec = NFRAG * 32 * 16 * 2 / 16;   // uv4 count = 2304
        uv4*       dst = (uv4*)wlds;
        const uv4* src = (const uv4*)wpack;
        for (int i = threadIdx.x; i < nvec; i += 256) dst[i] = src[i];
    }
    __syncthreads();

    const int tid  = threadIdx.x;
    const int wave = tid >> 5;
    const int lane = tid & 31;
    const int half = lane >> 4;
    const int col  = lane & 15;             // doubles as A-row m and C-col n
    bf16* hbuf = &hlds[wave * 16 * H_FEAT];

    // preload tiny invariants into registers (live across the tile loop)
    float biasl[8];
#pragma unroll
    for (int nt = 0; nt < 8; ++nt) biasl[nt] = b_l[nt * 16 + col];
    const float bias1 = b1[col];
    const float bias2 = b2[half];
    float w2c[16];
#pragma unroll
    for (int n = 0; n < 16; ++n) w2c[n] = W2[n * 2 + half];

    for (int tile = blockIdx.x * 8 + wave; tile < numTiles;
         tile += gridDim.x * 8) {
        // Prevent LICM from hoisting the per-iteration LDS B-fragment loads
        // out of the loop (previous round: 36 frags hoisted -> scratch spill).
        asm volatile("" ::: "memory");

        const int m0 = tile * 16;
        int arowi = m0 + col;
        if (arowi >= Nnodes) arowi = Nnodes - 1;
        const bf16* arow = acomb + (long long)arowi * H_FEAT;

        // ---- h = [agg|xn](16x128) @ W'(128x128) + b_l, f32 accumulate ----
        v8f c[8];
#pragma unroll
        for (int nt = 0; nt < 8; ++nt) {
            v8f t;
#pragma unroll
            for (int r = 0; r < 8; ++r) t[r] = biasl[nt];
            c[nt] = t;
        }
#pragma unroll
        for (int ks = 0; ks < 4; ++ks) {
            FragBF a;
            const int kb = ks * 32 + half * 16;
            a.u[0] = *(const uv4*)(arow + kb);
            a.u[1] = *(const uv4*)(arow + kb + 8);
#pragma unroll
            for (int nt = 0; nt < 8; ++nt) {
                FragBF b;
                const bf16* bp = wlds + (((nt * 4 + ks) * 32 + lane) << 4);
                b.u[0] = *(const uv4*)(bp);
                b.u[1] = *(const uv4*)(bp + 8);
                c[nt] = __builtin_amdgcn_wmma_f32_16x16x32_bf16(
                            false, a.v, false, b.v, (short)0, c[nt],
                            false, false);
            }
        }
        // relu -> bf16 -> LDS (transpose C-layout to row-major 16x128)
#pragma unroll
        for (int nt = 0; nt < 8; ++nt) {
#pragma unroll
            for (int r = 0; r < 8; ++r) {
                float v = c[nt][r];
                v = v > 0.0f ? v : 0.0f;
                hbuf[(r + half * 8) * H_FEAT + nt * 16 + col] = (bf16)v;
            }
        }
        // ---- s = relu(h)(16x128) @ W1(128x16) + b1 ----
        v8f s;
#pragma unroll
        for (int r = 0; r < 8; ++r) s[r] = bias1;
        const bf16* hrow = hbuf + col * H_FEAT;   // A row m = lane&15
#pragma unroll
        for (int ks = 0; ks < 4; ++ks) {
            FragBF a;
            const int kb = ks * 32 + half * 16;
            a.u[0] = *(const uv4*)(hrow + kb);
            a.u[1] = *(const uv4*)(hrow + kb + 8);
            FragBF b;
            const bf16* bp = wlds + (((32 + ks) * 32 + lane) << 4);
            b.u[0] = *(const uv4*)(bp);
            b.u[1] = *(const uv4*)(bp + 8);
            s = __builtin_amdgcn_wmma_f32_16x16x32_bf16(
                    false, a.v, false, b.v, (short)0, s, false, false);
        }
        // relu(s) -> LDS f32 (reuse wave-private buffer; same-wave DS in order)
        float* sbuf = (float*)hbuf;               // 16x16 f32 = 1KB
#pragma unroll
        for (int r = 0; r < 8; ++r) {
            float v = s[r];
            v = v > 0.0f ? v : 0.0f;
            sbuf[(r + half * 8) * 16 + col] = v;
        }
        // ---- out = relu(s) @ W2(16x2) + b2 : lane -> (m=col, c=half) ----
        {
            float acc = bias2;
#pragma unroll
            for (int n = 0; n < 16; ++n)
                acc += sbuf[col * 16 + n] * w2c[n];
            int mrow = m0 + col;
            if (mrow < Nnodes) out[(long long)mrow * 2 + half] = acc;
        }
    }
}

// ---------------------------------------------------------------------------
extern "C" void kernel_launch(void* const* d_in, const int* in_sizes, int n_in,
                              void* d_out, int out_size, void* d_ws,
                              size_t ws_size, hipStream_t stream) {
    const float*     x     = (const float*)d_in[0];
    const long long* ei    = (const long long*)d_in[1];
    // d_in[2..5]: edge_weight, edge_features, adj, T — unused by reference.
    const float*     gamma = (const float*)d_in[6];
    const float*     beta  = (const float*)d_in[7];
    const float*     Wl    = (const float*)d_in[8];
    const float*     bl    = (const float*)d_in[9];
    const float*     Wr    = (const float*)d_in[10];
    const float*     W1    = (const float*)d_in[11];
    const float*     b1    = (const float*)d_in[12];
    const float*     W2    = (const float*)d_in[13];
    const float*     b2    = (const float*)d_in[14];

    const int N = in_sizes[0] / D_FEAT;
    const int E = in_sizes[1] / 2;

    // ---- workspace layout (256B-aligned sections) ----
    char* ws = (char*)d_ws;
    size_t off = 0;
    float* sums   = (float*)(ws + off); off += 256;            // 64 f32
    float* sumsqs = (float*)(ws + off); off += 256;            // 64 f32
    float* scale  = (float*)(ws + off); off += 256;            // 64 f32
    float* shift  = (float*)(ws + off); off += 256;            // 64 f32
    size_t deg_b  = (((size_t)N * 4) + 255) & ~(size_t)255;
    float* deg    = (float*)(ws + off); off += deg_b;          // N f32
    size_t agg_b  = (size_t)N * D_FEAT * 4;
    float* agg    = (float*)(ws + off); off += agg_b;          // N*64 f32
    size_t ac_b   = (size_t)N * H_FEAT * 2;
    bf16* acomb   = (bf16*)(ws + off);  off += ac_b;           // N*128 bf16
    bf16* wpack   = (bf16*)(ws + off);  off += NFRAG * 32 * 16 * 2;

    // 1) zero stats + deg + agg (everything accumulated via atomics)
    long long zero_n = (long long)(1024 + deg_b + agg_b) / 4;
    zero_f32<<<4096, 256, 0, stream>>>((float*)ws, zero_n);

    // 2) BN statistics
    bn_stats<<<1024, 256, 0, stream>>>(x, sums, sumsqs, N);
    bn_finalize<<<1, 64, 0, stream>>>(sums, sumsqs, gamma, beta,
                                      scale, shift, N);

    // 3) BN apply -> bf16 xn half of combined A
    bn_apply<<<4096, 256, 0, stream>>>(x, scale, shift, acomb,
                                       (long long)N * D_FEAT);

    // 4) prepack weights into WMMA B-fragment layout
    pack_w<<<(NFRAG * 32 * 16 + 255) / 256, 256, 0, stream>>>(Wl, Wr, W1,
                                                              wpack);

    // 5) edge scatter (L2-resident f32 atomics) + degree
    edge_scatter<<<4096, 256, 0, stream>>>(ei, acomb, agg, deg, E);

    // 6) normalize agg -> bf16 agg half of combined A
    agg_normalize<<<4096, 256, 0, stream>>>(agg, deg, acomb,
                                            (long long)N * D_FEAT);

    // 7) fused WMMA GEMM + head
    int numTiles = (N + 15) / 16;
    int blocks = (numTiles + 7) / 8;
    if (blocks > 2048) blocks = 2048;
    gemm_fused<<<blocks, 256, 0, stream>>>(acomb, wpack, bl, b1, W2, b2,
                                           (float*)d_out, N, numTiles);
}
